// InfoNCELoss_75917841924443
// MI455X (gfx1250) — matrix operand order repaired and checked by reading, not compiled, not run on hardware
//
#include <hip/hip_runtime.h>

typedef __attribute__((ext_vector_type(16))) _Float16 v16h;
typedef __attribute__((ext_vector_type(8)))  float    v8f;

#define TAUINV    20.0f                 /* 1 / 0.05 */
#define INV2SIG2  0.8888888888888889f   /* 1 / (2 * 0.75^2) = 1/1.125 */
#define DIM       128
#define SUM_BLOCKS 256

// ---------------------------------------------------------------------------
// Reset the atomic-max scalars (harness does not re-poison ws between replays)
// ---------------------------------------------------------------------------
__global__ __launch_bounds__(64) void init_scalars(unsigned* m1, unsigned* m2) {
  if (threadIdx.x == 0 && blockIdx.x == 0) { *m1 = 0u; *m2 = 0u; }
}

// ---------------------------------------------------------------------------
// One wave per 16-edge tile.  For each tile, build f16 fragments of the
// per-edge difference vectors and accumulate D = diff x diff^T over K=128
// with 4 x v_wmma_f32_16x16x32_f16.  diag(D) = squared L2 distances.
// The 51.2 MB node table is L2-resident on MI455X (192 MB L2), so the random
// row gathers are served at L2 bandwidth.
// ---------------------------------------------------------------------------
__global__ __launch_bounds__(256) void edge_sim_wmma(
    const float* __restrict__ x,
    const int* __restrict__ att, const int* __restrict__ rep,
    float* __restrict__ sim1, float* __restrict__ sim2,
    unsigned* __restrict__ max1bits, unsigned* __restrict__ max2bits,
    int eAtt, int eRep)
{
  const int lane = threadIdx.x & 31;
  const int wid  = blockIdx.x * (blockDim.x >> 5) + (threadIdx.x >> 5);
  const int tilesAtt = eAtt >> 4;
  const int tilesTot = (eAtt + eRep) >> 4;
  if (wid >= tilesTot) return;          // wave-uniform: EXEC stays all-ones

  const bool isAtt = wid < tilesAtt;
  const int  tile  = isAtt ? wid : wid - tilesAtt;
  const int  E     = isAtt ? eAtt : eRep;
  const int* __restrict__ edges  = isAtt ? att : rep;
  float*     __restrict__ simArr = isAtt ? sim1 : sim2;
  unsigned*               maxPtr = isAtt ? max1bits : max2bits;

  const int col = lane & 15;            // edge within the tile (A row / B col)
  const int hf  = lane >> 4;            // which half-wave (K-chunk split)
  const int e   = (tile << 4) + col;
  const int eu  = edges[e];
  const int ev  = edges[E + e];
  const float4* __restrict__ u4 = (const float4*)(x + (size_t)eu * DIM);
  const float4* __restrict__ v4 = (const float4*)(x + (size_t)ev * DIM);

  v8f c = {};
#pragma unroll
  for (int kc = 0; kc < 4; ++kc) {      // K = 128 in 4 chunks of 32
    // This lane needs diff elements [kbase + 8*hf .. +24): 24 contiguous f32
    const int fb = kc * 8 + hf * 2;     // float4 index into the row
    float d[24];
#pragma unroll
    for (int j = 0; j < 6; ++j) {
      const float4 ua = u4[fb + j];
      const float4 va = v4[fb + j];
      d[4 * j + 0] = ua.x - va.x;
      d[4 * j + 1] = ua.y - va.y;
      d[4 * j + 2] = ua.z - va.z;
      d[4 * j + 3] = ua.w - va.w;
    }
    // A fragment (16x32 f16): lane holds M=col, K = local {0..7} u {16..23}
    v16h af, bf;
#pragma unroll
    for (int j = 0; j < 8; ++j) {
      af[j]     = (_Float16)d[j];
      af[j + 8] = (_Float16)d[j + 16];
    }
    // B fragment (32x16 f16): lane holds N=col, K = local {0..7} u {16..23}
#pragma unroll
    for (int j = 0; j < 16; ++j) {
      const float s = (j < 8) ? d[j] : d[j + 8];   // constant-index selects
      bf[j] = (_Float16)s;
    }
    c = __builtin_amdgcn_wmma_f32_16x16x32_f16(false, af, false, bf,
                                               (short)0, c, false, false);
  }

  // Diagonal of C: edge i lives on lane i (i<8, VGPR i) or lane i+16 (i>=8,
  // VGPR lane-24).  Those lanes already hold edge i's data -> no shuffle.
  const int rr = lane & 7;
  float l2sq =
      (rr == 0) ? c[0] : (rr == 1) ? c[1] : (rr == 2) ? c[2] : (rr == 3) ? c[3]
    : (rr == 4) ? c[4] : (rr == 5) ? c[5] : (rr == 6) ? c[6] : c[7];

  const bool holder = (lane < 8) || (lane >= 24);
  float t = 0.0f;                        // sims > 0, so 0 is a safe identity
  if (holder) {
    const float l2  = sqrtf(fmaxf(l2sq, 0.0f));
    const float sim = __expf(-l2 * INV2SIG2);
    simArr[(tile << 4) + col] = sim;
    t = sim * TAUINV;
  }
#pragma unroll
  for (int off = 16; off > 0; off >>= 1)
    t = fmaxf(t, __shfl_xor(t, off, 32));
  if (lane == 0) atomicMax(maxPtr, __float_as_uint(t));  // positive-float bits
}

// ---------------------------------------------------------------------------
// Deterministic sum of exp(sim2/TAU - max2): fixed-grid block partials ...
// ---------------------------------------------------------------------------
__global__ __launch_bounds__(256) void rep_sum_partial(
    const float* __restrict__ sim2, const unsigned* __restrict__ max2bits,
    float* __restrict__ partials, int eRep)
{
  __shared__ float sh[256];
  const float max2 = __uint_as_float(*max2bits);
  float s = 0.0f;
  for (int i = blockIdx.x * blockDim.x + threadIdx.x; i < eRep;
       i += gridDim.x * blockDim.x)
    s += __expf(sim2[i] * TAUINV - max2);
  sh[threadIdx.x] = s;
  __syncthreads();
#pragma unroll
  for (int st = 128; st > 0; st >>= 1) {
    if (threadIdx.x < st) sh[threadIdx.x] += sh[threadIdx.x + st];
    __syncthreads();
  }
  if (threadIdx.x == 0) partials[blockIdx.x] = sh[0];
}

// ... then a single-block fixed-order tree reduce.
__global__ __launch_bounds__(256) void rep_sum_final(
    const float* __restrict__ partials, float* __restrict__ s2out)
{
  __shared__ float sh[SUM_BLOCKS];
  sh[threadIdx.x] = partials[threadIdx.x];
  __syncthreads();
#pragma unroll
  for (int st = SUM_BLOCKS / 2; st > 0; st >>= 1) {
    if (threadIdx.x < st) sh[threadIdx.x] += sh[threadIdx.x + st];
    __syncthreads();
  }
  if (threadIdx.x == 0) *s2out = sh[0];
}

// ---------------------------------------------------------------------------
// out[i] = -log(num/denom) = log(exp(t1-max1) + S2) - (t1 - max1)
// ---------------------------------------------------------------------------
__global__ __launch_bounds__(256) void att_loss(
    const float* __restrict__ sim1, const unsigned* __restrict__ max1bits,
    const float* __restrict__ s2ptr, float* __restrict__ out, int eAtt)
{
  const int i = blockIdx.x * blockDim.x + threadIdx.x;
  if (i >= eAtt) return;
  const float max1 = __uint_as_float(*max1bits);
  const float S2   = *s2ptr;
  const float a    = sim1[i] * TAUINV - max1;   // t1 - max1  (<= 0)
  const float num  = __expf(a);
  out[i] = __logf(num + S2) - a;
}

// ---------------------------------------------------------------------------
extern "C" void kernel_launch(void* const* d_in, const int* in_sizes, int n_in,
                              void* d_out, int out_size, void* d_ws, size_t ws_size,
                              hipStream_t stream) {
  const float* x   = (const float*)d_in[0];
  const int*   att = (const int*)d_in[1];
  const int*   rep = (const int*)d_in[2];
  const int eAtt = in_sizes[1] / 2;   // 300000
  const int eRep = in_sizes[2] / 2;   // 600000
  float* out = (float*)d_out;

  // ws layout (floats): sim1 | sim2 | max1bits | max2bits | S2 | partials[256]
  float*    ws       = (float*)d_ws;
  float*    sim1     = ws;
  float*    sim2     = ws + eAtt;
  unsigned* max1bits = (unsigned*)(ws + eAtt + eRep);
  unsigned* max2bits = max1bits + 1;
  float*    s2ptr    = ws + eAtt + eRep + 2;
  float*    partials = ws + eAtt + eRep + 3;

  init_scalars<<<1, 64, 0, stream>>>(max1bits, max2bits);

  const int tilesTot = (eAtt + eRep) / 16;            // 56250 waves
  const int blocksA  = (tilesTot + 7) / 8;            // 8 waves / 256-thread block
  edge_sim_wmma<<<blocksA, 256, 0, stream>>>(x, att, rep, sim1, sim2,
                                             max1bits, max2bits, eAtt, eRep);

  rep_sum_partial<<<SUM_BLOCKS, 256, 0, stream>>>(sim2, max2bits, partials, eRep);
  rep_sum_final<<<1, SUM_BLOCKS, 0, stream>>>(partials, s2ptr);

  att_loss<<<(eAtt + 255) / 256, 256, 0, stream>>>(sim1, max1bits, s2ptr, out, eAtt);
}
